// VAE_17059610100167
// MI455X (gfx1250) — compile-verified, matrix operand order checked by reading, not run on hardware
//
#include <hip/hip_runtime.h>
#include <hip/hip_bf16.h>
#include <math.h>

typedef __attribute__((ext_vector_type(8)))  float   v8f;
typedef __attribute__((ext_vector_type(2)))  __bf16  v2bf;
typedef __attribute__((ext_vector_type(4)))  __bf16  v4bf;
typedef __attribute__((ext_vector_type(8)))  __bf16  v8bf;
typedef __attribute__((ext_vector_type(16))) __bf16  v16bf;

#define BSZ   4096
#define NODE  6
#define ND    2
#define LATD  12
#define IMGD  27648
#define BN_EPS 1e-5f
#define KSPLIT 8
#define KCHUNK (IMGD / KSPLIT)     /* 3456, multiple of 32 */

// ---------------- output layout (floats) ----------------
#define OUT_MEAN   0
#define OUT_LOGVAR (OUT_MEAN   + BSZ*LATD)
#define OUT_PRIOR  (OUT_LOGVAR + BSZ*LATD)
#define OUT_L1     (OUT_PRIOR  + BSZ*LATD)
#define OUT_L2     (OUT_L1     + BSZ*LATD)
#define OUT_L3     (OUT_L2     + BSZ*LATD)
#define OUT_ALIGN  (OUT_L3     + BSZ*LATD)
#define OUT_XHAT   (OUT_ALIGN  + BSZ*NODE)

__device__ __forceinline__ float eluf(float x)  { return x > 0.f ? x : expf(x) - 1.f; }

__device__ __forceinline__ v16bf cat16(v8bf lo, v8bf hi) {
    return __builtin_shufflevector(lo, hi, 0,1,2,3,4,5,6,7,8,9,10,11,12,13,14,15);
}

// =====================================================================
// Generic GEMM: C[M x N] = act(A[M x K] @ W[K x N] + bias), f32 in/out,
// bf16 WMMA core. M must be a multiple of 128. ACT: 0 none, 1 elu, 2 tanh
// Block: 256 thr = 8 waves, tile 128(M) x 64(N), waves 4(M) x 2(N).
// Used where A/W are L2-resident (K=300 layers, dec_W3 layer).
// =====================================================================
template<int ACT>
__global__ __launch_bounds__(256)
void gemm_wmma_bf16(const float* __restrict__ A, const float* __restrict__ W,
                    const float* __restrict__ bias, float* __restrict__ C,
                    int M, int K, int N)
{
    __shared__ __align__(16) __bf16 lA[128 * 32];     // [m][k]
    __shared__ __align__(16) __bf16 lB[64 * 32];      // [n][k]  (transposed)

    const int tid   = threadIdx.x;
    const int lane  = tid & 31;
    const int wave  = tid >> 5;
    const int wm    = wave & 3;
    const int wn    = wave >> 2;
    const int l16   = lane & 15;
    const int half  = lane >> 4;

    const int mBase = blockIdx.x * 128;
    const int nBase = blockIdx.y * 64;

    v8f acc[2][2];
    acc[0][0] = {}; acc[0][1] = {}; acc[1][0] = {}; acc[1][1] = {};

    for (int k0 = 0; k0 < K; k0 += 32) {
        const bool fullK = (k0 + 32 <= K);          // wave-uniform
        const bool fullN = (nBase + 64 <= N);       // wave-uniform

        // ---- stage A tile (f32 -> bf16) ----
        if (fullK) {
            #pragma unroll
            for (int it = 0; it < 4; ++it) {         // 1024 float4 / 256 thr
                int idx4 = tid + it * 256;
                int r = idx4 >> 3, c4 = (idx4 & 7) << 2;
                const float4 v = *(const float4*)&A[(size_t)(mBase + r) * K + k0 + c4];
                v4bf b; b[0] = (__bf16)v.x; b[1] = (__bf16)v.y; b[2] = (__bf16)v.z; b[3] = (__bf16)v.w;
                *(v4bf*)&lA[r * 32 + c4] = b;
                if (k0 + 64 <= K)
                    __builtin_prefetch(&A[(size_t)(mBase + r) * K + k0 + 32 + c4], 0, 1);
            }
        } else {
            for (int idx = tid; idx < 128 * 32; idx += 256) {   // branchless tail
                int r = idx >> 5, c = idx & 31;
                int gc = k0 + c, cc = gc < K ? gc : K - 1;
                float v = A[(size_t)(mBase + r) * K + cc];
                lA[idx] = (__bf16)(gc < K ? v : 0.f);
            }
        }
        // ---- stage B tile transposed (f32 -> bf16) ----
        if (fullK && fullN) {
            #pragma unroll
            for (int it = 0; it < 4; ++it) {         // 64n x 16 k-pairs / 256 thr
                int idx = tid + it * 256;
                int n = idx & 63, k2 = idx >> 6;
                int gn = nBase + n, gk = k0 + 2 * k2;
                float a0 = W[(size_t)gk * N + gn];
                float a1 = W[(size_t)(gk + 1) * N + gn];
                v2bf b; b[0] = (__bf16)a0; b[1] = (__bf16)a1;
                *(v2bf*)&lB[n * 32 + 2 * k2] = b;
            }
        } else {
            for (int idx = tid; idx < 64 * 32; idx += 256) {    // branchless edges
                int n = idx & 63, k = idx >> 6;
                int gn = nBase + n, gk = k0 + k;
                int cg = gn < N ? gn : N - 1;
                int kg = gk < K ? gk : K - 1;
                float v = W[(size_t)kg * N + cg];
                lB[n * 32 + k] = (__bf16)((gn < N && gk < K) ? v : 0.f);
            }
        }
        __syncthreads();

        v16bf afr[2], bfr[2];
        #pragma unroll
        for (int mi = 0; mi < 2; ++mi) {
            const __bf16* p = &lA[(wm * 32 + mi * 16 + l16) * 32];
            afr[mi] = cat16(*(const v8bf*)(p + half * 8), *(const v8bf*)(p + 16 + half * 8));
        }
        #pragma unroll
        for (int ni = 0; ni < 2; ++ni) {
            const __bf16* p = &lB[(wn * 32 + ni * 16 + l16) * 32];
            bfr[ni] = cat16(*(const v8bf*)(p + half * 8), *(const v8bf*)(p + 16 + half * 8));
        }
        #pragma unroll
        for (int mi = 0; mi < 2; ++mi)
            #pragma unroll
            for (int ni = 0; ni < 2; ++ni)
                acc[mi][ni] = __builtin_amdgcn_wmma_f32_16x16x32_bf16(
                    false, afr[mi], false, bfr[ni], (short)0, acc[mi][ni], false, false);
        __syncthreads();
    }

    #pragma unroll
    for (int mi = 0; mi < 2; ++mi) {
        #pragma unroll
        for (int ni = 0; ni < 2; ++ni) {
            int gn = nBase + wn * 32 + ni * 16 + l16;
            if (gn >= N) continue;
            float bv = bias[gn];
            #pragma unroll
            for (int v = 0; v < 8; ++v) {
                int gm = mBase + wm * 32 + mi * 16 + half * 8 + v;
                float x = acc[mi][ni][v] + bv;
                if (ACT == 1) x = eluf(x);
                else if (ACT == 2) x = tanhf(x);
                C[(size_t)gm * N + gn] = x;
            }
        }
    }
}

// =====================================================================
// GEMM1 split-K: partial[kc] = image[128-row tile, K-chunk] @ enc_W1.
// One block covers ALL of N (300, padded to 320) so the 453 MB image is
// streamed from HBM exactly once. 512 thr = 16 waves, waves 4(M) x 4(N),
// wave tile 32 x 80 (2 x 5 WMMA tiles). grid = (M/128, KSPLIT).
// =====================================================================
__global__ __launch_bounds__(512)
void gemm1_splitk(const float* __restrict__ A, const float* __restrict__ W,
                  float* __restrict__ part)
{
    __shared__ __align__(16) __bf16 lA[128 * 32];     // [m][k]
    __shared__ __align__(16) __bf16 lB[320 * 32];     // [n][k] (transposed)

    const int tid   = threadIdx.x;
    const int lane  = tid & 31;
    const int wave  = tid >> 5;           // 0..15
    const int wm    = wave & 3;
    const int wn    = wave >> 2;          // 0..3, covers 80 cols each
    const int l16   = lane & 15;
    const int half  = lane >> 4;

    const int mBase = blockIdx.x * 128;
    const int kc    = blockIdx.y;
    const int ks    = kc * KCHUNK;
    const int N = 300, K = IMGD;

    v8f acc[2][5];
    #pragma unroll
    for (int mi = 0; mi < 2; ++mi)
        #pragma unroll
        for (int ni = 0; ni < 5; ++ni) acc[mi][ni] = {};

    for (int k0 = ks; k0 < ks + KCHUNK; k0 += 32) {
        // ---- stage A (always aligned): 1024 float4 / 512 thr ----
        #pragma unroll
        for (int it = 0; it < 2; ++it) {
            int idx4 = tid + it * 512;
            int r = idx4 >> 3, c4 = (idx4 & 7) << 2;
            const float4 v = *(const float4*)&A[(size_t)(mBase + r) * K + k0 + c4];
            v4bf b; b[0] = (__bf16)v.x; b[1] = (__bf16)v.y; b[2] = (__bf16)v.z; b[3] = (__bf16)v.w;
            *(v4bf*)&lA[r * 32 + c4] = b;
            if (k0 + 64 <= ks + KCHUNK)
                __builtin_prefetch(&A[(size_t)(mBase + r) * K + k0 + 32 + c4], 0, 1);
        }
        // ---- stage B transposed, branchless n-clamp: 320n x 16 k-pairs ----
        for (int idx = tid; idx < 320 * 16; idx += 512) {
            int n = idx % 320, k2 = idx / 320;
            int gk = k0 + 2 * k2;
            int cg = n < N ? n : N - 1;
            float a0 = W[(size_t)gk * N + cg];
            float a1 = W[(size_t)(gk + 1) * N + cg];
            v2bf b;
            b[0] = (__bf16)(n < N ? a0 : 0.f);
            b[1] = (__bf16)(n < N ? a1 : 0.f);
            *(v2bf*)&lB[n * 32 + 2 * k2] = b;
        }
        __syncthreads();

        v16bf afr[2];
        #pragma unroll
        for (int mi = 0; mi < 2; ++mi) {
            const __bf16* p = &lA[(wm * 32 + mi * 16 + l16) * 32];
            afr[mi] = cat16(*(const v8bf*)(p + half * 8), *(const v8bf*)(p + 16 + half * 8));
        }
        #pragma unroll
        for (int ni = 0; ni < 5; ++ni) {
            const __bf16* p = &lB[(wn * 80 + ni * 16 + l16) * 32];
            v16bf bfr = cat16(*(const v8bf*)(p + half * 8), *(const v8bf*)(p + 16 + half * 8));
            #pragma unroll
            for (int mi = 0; mi < 2; ++mi)
                acc[mi][ni] = __builtin_amdgcn_wmma_f32_16x16x32_bf16(
                    false, afr[mi], false, bfr, (short)0, acc[mi][ni], false, false);
        }
        __syncthreads();
    }

    #pragma unroll
    for (int mi = 0; mi < 2; ++mi) {
        #pragma unroll
        for (int ni = 0; ni < 5; ++ni) {
            int gn = wn * 80 + ni * 16 + l16;
            if (gn >= N) continue;
            #pragma unroll
            for (int v = 0; v < 8; ++v) {
                int gm = mBase + wm * 32 + mi * 16 + half * 8 + v;
                part[((size_t)kc * BSZ + gm) * N + gn] = acc[mi][ni][v];
            }
        }
    }
}

__global__ void reduce_splitk(const float* __restrict__ part, const float* __restrict__ bias,
                              float* __restrict__ h1) {
    int i = blockIdx.x * blockDim.x + threadIdx.x;
    if (i >= BSZ * 300) return;
    float s = 0.f;
    #pragma unroll
    for (int kc = 0; kc < KSPLIT; ++kc) s += part[(size_t)kc * BSZ * 300 + i];
    h1[i] = eluf(s + bias[i % 300]);
}

// =====================================================================
// Small helper kernels (unchanged)
// =====================================================================
__global__ void zero_f32(float* p, int n) {
    int i = blockIdx.x * blockDim.x + threadIdx.x;
    if (i < n) p[i] = 0.f;
}

__global__ __launch_bounds__(256)
void bn_stats(const float* __restrict__ h3, float* __restrict__ stats) {
    int c = blockIdx.x;
    __shared__ float s[256], q[256];
    float sum = 0.f, sq = 0.f;
    for (int r = threadIdx.x; r < BSZ; r += 256) {
        float v = h3[r * 24 + c];
        sum += v; sq += v * v;
    }
    s[threadIdx.x] = sum; q[threadIdx.x] = sq;
    __syncthreads();
    for (int st = 128; st > 0; st >>= 1) {
        if (threadIdx.x < st) { s[threadIdx.x] += s[threadIdx.x + st]; q[threadIdx.x] += q[threadIdx.x + st]; }
        __syncthreads();
    }
    if (threadIdx.x == 0) {
        float m = s[0] / (float)BSZ;
        stats[c]      = m;
        stats[24 + c] = q[0] / (float)BSZ - m * m;
    }
}

__global__ void bn_apply_eps(const float* __restrict__ h3, const float* __restrict__ stats,
                             const float* __restrict__ g, const float* __restrict__ be,
                             const float* __restrict__ noise,
                             float* __restrict__ out, float* __restrict__ eps) {
    int i = blockIdx.x * blockDim.x + threadIdx.x;
    if (i >= BSZ * LATD) return;
    int r = i / LATD, l = i % LATD;
    float hm = h3[r * 24 + l], hl = h3[r * 24 + 12 + l];
    float ym = (hm - stats[l])      * rsqrtf(stats[24 + l]      + BN_EPS) * g[l]      + be[l];
    float yl = (hl - stats[12 + l]) * rsqrtf(stats[24 + 12 + l] + BN_EPS) * g[12 + l] + be[12 + l];
    out[OUT_MEAN + i]   = ym;
    out[OUT_LOGVAR + i] = yl;
    eps[i] = ym + expf(0.5f * yl) * noise[i];
}

__global__ __launch_bounds__(1024)
void lgp_node(int j,
              const float* __restrict__ Bm,  const float* __restrict__ Wa,
              const float* __restrict__ ba,  const float* __restrict__ Wb,
              const float* __restrict__ bb,  const float* __restrict__ Wc,
              const float* __restrict__ bc,  const float* __restrict__ g,
              const float* __restrict__ be,  const float* __restrict__ alW,
              const float* __restrict__ alb, const float* __restrict__ eps,
              float* __restrict__ lat2, float* __restrict__ out)
{
    __shared__ float rs[1024], r1[1024], r2[1024], r3[1024];
    __shared__ float stat[4];
    const int t = threadIdx.x;

    float mask[LATD];
    #pragma unroll
    for (int i = 0; i < LATD; ++i) mask[i] = Bm[(i % NODE) * NODE + j];

    float h0[4], h1v[4];
    float s0 = 0.f, s1 = 0.f, q0 = 0.f, q1 = 0.f;
    #pragma unroll
    for (int p = 0; p < 4; ++p) {
        int r = t + p * 1024;
        const float* l2 = &lat2[r * LATD];
        float a[NODE];
        #pragma unroll
        for (int k = 0; k < NODE; ++k) a[k] = ba[j * NODE + k];
        #pragma unroll
        for (int i = 0; i < LATD; ++i) {
            float ml = mask[i] * l2[i];
            #pragma unroll
            for (int k = 0; k < NODE; ++k) a[k] += ml * Wa[j * LATD * NODE + i * NODE + k];
        }
        #pragma unroll
        for (int k = 0; k < NODE; ++k) a[k] = eluf(a[k]);
        float b0 = bb[j * 2 + 0], b1 = bb[j * 2 + 1];
        #pragma unroll
        for (int k = 0; k < NODE; ++k) {
            b0 += a[k] * Wb[j * NODE * 2 + k * 2 + 0];
            b1 += a[k] * Wb[j * NODE * 2 + k * 2 + 1];
        }
        b0 = eluf(b0); b1 = eluf(b1);
        float c0 = bc[j * 2 + 0] + b0 * Wc[j * 4 + 0] + b1 * Wc[j * 4 + 2];
        float c1 = bc[j * 2 + 1] + b0 * Wc[j * 4 + 1] + b1 * Wc[j * 4 + 3];
        h0[p] = c0; h1v[p] = c1;
        s0 += c0; s1 += c1; q0 += c0 * c0; q1 += c1 * c1;
    }
    rs[t] = s0; r1[t] = s1; r2[t] = q0; r3[t] = q1;
    __syncthreads();
    for (int st = 512; st > 0; st >>= 1) {
        if (t < st) { rs[t] += rs[t+st]; r1[t] += r1[t+st]; r2[t] += r2[t+st]; r3[t] += r3[t+st]; }
        __syncthreads();
    }
    if (t == 0) {
        float m0 = rs[0] / (float)BSZ, m1 = r1[0] / (float)BSZ;
        stat[0] = m0; stat[1] = m1;
        stat[2] = rsqrtf(r2[0] / (float)BSZ - m0 * m0 + BN_EPS);
        stat[3] = rsqrtf(r3[0] / (float)BSZ - m1 * m1 + BN_EPS);
    }
    __syncthreads();
    const float m0 = stat[0], m1 = stat[1], rstd0 = stat[2], rstd1 = stat[3];
    const float g0 = g[j*2], g1 = g[j*2+1], be0 = be[j*2], be1 = be[j*2+1];
    const float aw0 = alW[j*2], aw1 = alW[j*2+1], ab = alb[j];
    #pragma unroll
    for (int p = 0; p < 4; ++p) {
        int r = t + p * 1024;
        float y0 = (h0[p]  - m0) * rstd0 * g0 + be0;
        float y1 = (h1v[p] - m1) * rstd1 * g1 + be1;
        out[OUT_L1 + r * LATD + j]        = y0;
        out[OUT_L1 + r * LATD + j + NODE] = y1;
        float z0 = y0 + eps[r * LATD + j];
        float z1 = y1 + eps[r * LATD + j + NODE];
        lat2[r * LATD + j]        = z0;
        lat2[r * LATD + j + NODE] = z1;
        out[OUT_ALIGN + r * NODE + j] = z0 * aw0 + z1 * aw1 + ab;
    }
}

__global__ void lat_finalize(const float* __restrict__ lat2, float* __restrict__ lat3,
                             float* __restrict__ out) {
    int i = blockIdx.x * blockDim.x + threadIdx.x;
    if (i >= BSZ * LATD) return;
    float z = lat2[i];
    float t3 = tanhf(z);
    out[OUT_L2 + i] = z;
    out[OUT_L3 + i] = t3;
    lat3[i] = t3;
}

__global__ void dec1_kernel(const float* __restrict__ lat3, const float* __restrict__ W,
                            const float* __restrict__ b, float* __restrict__ d1) {
    int i = blockIdx.x * blockDim.x + threadIdx.x;
    if (i >= BSZ * 300) return;
    int r = i / 300, c = i % 300;
    float acc = b[c];
    const float* l = &lat3[r * LATD];
    #pragma unroll
    for (int k = 0; k < LATD; ++k) acc += l[k] * W[k * 300 + c];
    d1[i] = eluf(acc);
}

__global__ void prior_kernel(const float* __restrict__ label, const float* __restrict__ W1,
                             const float* __restrict__ b1, const float* __restrict__ W2,
                             const float* __restrict__ b2, float* __restrict__ out) {
    int i = blockIdx.x * blockDim.x + threadIdx.x;
    if (i >= BSZ * NODE) return;
    int r = i / NODE, n = i % NODE;
    float x = label[i];
    float ph[3];
    #pragma unroll
    for (int h = 0; h < 3; ++h) ph[h] = eluf(x * W1[n * 3 + h] + b1[n * 3 + h]);
    #pragma unroll
    for (int d = 0; d < ND; ++d) {
        float acc = b2[n * ND + d];
        #pragma unroll
        for (int h = 0; h < 3; ++h) acc += ph[h] * W2[n * 3 * ND + h * ND + d];
        out[OUT_PRIOR + r * LATD + n * ND + d] = acc;
    }
}

// =====================================================================
extern "C" void kernel_launch(void* const* d_in, const int* in_sizes, int n_in,
                              void* d_out, int out_size, void* d_ws, size_t ws_size,
                              hipStream_t stream) {
    const float* image  = (const float*)d_in[0];
    const float* label  = (const float*)d_in[1];
    const float* noise  = (const float*)d_in[2];
    const float* Bm     = (const float*)d_in[3];
    const float* enc_W1 = (const float*)d_in[4];
    const float* enc_b1 = (const float*)d_in[5];
    const float* enc_W2 = (const float*)d_in[6];
    const float* enc_b2 = (const float*)d_in[7];
    const float* enc_W3 = (const float*)d_in[8];
    const float* enc_b3 = (const float*)d_in[9];
    const float* enc_g  = (const float*)d_in[10];
    const float* enc_be = (const float*)d_in[11];
    const float* lgp_Wa = (const float*)d_in[12];
    const float* lgp_ba = (const float*)d_in[13];
    const float* lgp_Wb = (const float*)d_in[14];
    const float* lgp_bb = (const float*)d_in[15];
    const float* lgp_Wc = (const float*)d_in[16];
    const float* lgp_bc = (const float*)d_in[17];
    const float* lgp_g  = (const float*)d_in[18];
    const float* lgp_be = (const float*)d_in[19];
    const float* alW    = (const float*)d_in[20];
    const float* alb    = (const float*)d_in[21];
    const float* dec_W1 = (const float*)d_in[22];
    const float* dec_b1 = (const float*)d_in[23];
    const float* dec_W2 = (const float*)d_in[24];
    const float* dec_b2 = (const float*)d_in[25];
    const float* dec_W3 = (const float*)d_in[26];
    const float* dec_b3 = (const float*)d_in[27];
    const float* pr_W1  = (const float*)d_in[28];
    const float* pr_b1  = (const float*)d_in[29];
    const float* pr_W2  = (const float*)d_in[30];
    const float* pr_b2  = (const float*)d_in[31];

    float* out = (float*)d_out;

    // workspace carve-up (floats)
    float* ws   = (float*)d_ws;
    size_t o = 0;
    float* part = ws + o; o += (size_t)KSPLIT * BSZ * 300;   // 39.3 MB split-K partials
    float* h1   = ws + o; o += (size_t)BSZ * 300;
    float* h2   = ws + o; o += (size_t)BSZ * 300;
    float* d1   = ws + o; o += (size_t)BSZ * 300;
    float* d2   = ws + o; o += (size_t)BSZ * 300;
    float* h3   = ws + o; o += (size_t)BSZ * 24;
    float* eps  = ws + o; o += (size_t)BSZ * LATD;
    float* lat2 = ws + o; o += (size_t)BSZ * LATD;
    float* lat3 = ws + o; o += (size_t)BSZ * LATD;
    float* stats= ws + o; o += 64;

    const dim3 blk256(256);

    // ---- encoder ----
    gemm1_splitk<<<dim3(BSZ/128, KSPLIT), dim3(512), 0, stream>>>(image, enc_W1, part);
    reduce_splitk<<<(BSZ*300 + 255)/256, blk256, 0, stream>>>(part, enc_b1, h1);
    gemm_wmma_bf16<1><<<dim3(BSZ/128, (300 + 63)/64), blk256, 0, stream>>>(h1, enc_W2, enc_b2, h2, BSZ, 300, 300);
    gemm_wmma_bf16<0><<<dim3(BSZ/128, 1),             blk256, 0, stream>>>(h2, enc_W3, enc_b3, h3, BSZ, 300, 24);
    bn_stats<<<24, blk256, 0, stream>>>(h3, stats);
    bn_apply_eps<<<(BSZ*LATD + 255)/256, blk256, 0, stream>>>(h3, stats, enc_g, enc_be, noise, out, eps);

    // ---- causal latent graph (sequential over nodes) ----
    zero_f32<<<(BSZ*LATD + 255)/256, blk256, 0, stream>>>(lat2, BSZ*LATD);
    for (int j = 0; j < NODE; ++j) {
        lgp_node<<<1, 1024, 0, stream>>>(j, Bm, lgp_Wa, lgp_ba, lgp_Wb, lgp_bb,
                                         lgp_Wc, lgp_bc, lgp_g, lgp_be, alW, alb,
                                         eps, lat2, out);
    }
    lat_finalize<<<(BSZ*LATD + 255)/256, blk256, 0, stream>>>(lat2, lat3, out);

    // ---- decoder ----
    dec1_kernel<<<(BSZ*300 + 255)/256, blk256, 0, stream>>>(lat3, dec_W1, dec_b1, d1);
    gemm_wmma_bf16<1><<<dim3(BSZ/128, (300 + 63)/64),  blk256, 0, stream>>>(d1, dec_W2, dec_b2, d2, BSZ, 300, 300);
    gemm_wmma_bf16<2><<<dim3(BSZ/128, (IMGD + 63)/64), blk256, 0, stream>>>(d2, dec_W3, dec_b3, out + OUT_XHAT, BSZ, 300, IMGD);

    // ---- prior ----
    prior_kernel<<<(BSZ*NODE + 255)/256, blk256, 0, stream>>>(label, pr_W1, pr_b1, pr_W2, pr_b2, out);
}
